// FP_Lite_16123307229738
// MI455X (gfx1250) — matrix operand order repaired and checked by reading, not compiled, not run on hardware
//
#include <hip/hip_runtime.h>

// ---------------------------------------------------------------------------
// PointNet++ feature propagation, fused for MI455X (gfx1250, wave32, WMMA).
// One block = (batch, 64-row n-tile), 512 threads = 16 waves.
// Each wave owns one 16-col tile and reuses its B fragment across 4 row-tiles
// (cuts interp-GEMM L2 traffic 4x vs the 16-row version).
// Matmuls use v_wmma_f32_16x16x32_bf16 (fp32 accumulate).
// ---------------------------------------------------------------------------

typedef __attribute__((ext_vector_type(16))) __bf16 v16bf;
typedef __attribute__((ext_vector_type(8)))  __bf16 v8bf;
typedef __attribute__((ext_vector_type(8)))  float  v8f;

static constexpr int Bsz = 16;
static constexpr int Nn  = 4096;
static constexpr int Mm  = 1024;
static constexpr int C1  = 128;
static constexpr int C2  = 256;
static constexpr int CIN = 384;   // C1 + C2
static constexpr int H1  = 256;
static constexpr int H2  = 128;
static constexpr int RT  = 64;    // rows per block (4 row-tiles of 16)

// dynamic LDS layout (bytes)
static constexpr int LDS_WLS  = 0;                       // 64x1024 bf16 = 131072
static constexpr int LDS_NPL  = 131072;                  // 64x384  bf16 = 49152
static constexpr int LDS_HLS  = 131072 + 49152;          // 64x256  bf16 = 32768
static constexpr int LDS_PSUM = 131072 + 49152 + 32768;  // 512 f32      = 2048
static constexpr int LDS_RINV = LDS_PSUM + 2048;         // 64 f32       = 256
static constexpr int LDS_TOTAL = LDS_RINV + 256;         // 215296 B (< 320KB/WG)

__device__ __forceinline__ unsigned short f2bf(float x) {
  unsigned int u = __builtin_bit_cast(unsigned int, x);
  u += 0x7FFFu + ((u >> 16) & 1u);          // round-to-nearest-even
  return (unsigned short)(u >> 16);
}

// generic fp32 -> bf16 conversion (points2, W1, W2)
__global__ void cvt_f32_bf16(const float* __restrict__ in,
                             unsigned short* __restrict__ out, int n) {
  int i = blockIdx.x * blockDim.x + threadIdx.x;
  if (i < n) out[i] = f2bf(in[i]);
}

__global__ __launch_bounds__(512)
void fused_fp_kernel(const float* __restrict__ xyz1,
                     const float* __restrict__ xyz2,
                     const float* __restrict__ points1,
                     const unsigned short* __restrict__ p2bf,   // [B][C2][M] bf16
                     const unsigned short* __restrict__ w1bf,   // [H1][CIN] bf16
                     const float* __restrict__ b1,
                     const unsigned short* __restrict__ w2bf,   // [H2][H1] bf16
                     const float* __restrict__ b2,
                     float* __restrict__ out) {                 // [B][H2][N]
  extern __shared__ __align__(128) char smem[];
  unsigned short* wls = (unsigned short*)(smem + LDS_WLS);   // bf16 inv-dist
  unsigned short* npl = (unsigned short*)(smem + LDS_NPL);   // bf16 new_points
  unsigned short* hls = (unsigned short*)(smem + LDS_HLS);   // bf16 hidden
  float* psum   = (float*)(smem + LDS_PSUM);
  float* rowinv = (float*)(smem + LDS_RINV);

  const int t  = threadIdx.x;
  const int b  = blockIdx.x >> 6;            // 64 n-tiles (of 64 rows) per batch
  const int n0 = (blockIdx.x & 63) * RT;

  // ---- Phase 1: inverse distances for 64 query rows vs all M source pts ----
  {
    const int r = t >> 3;                    // 8 threads per row, 64 rows
    const int i = t & 7;
    const float* p1 = xyz1 + (size_t)(b * Nn + n0 + r) * 3;
    const float x1 = p1[0], y1 = p1[1], z1 = p1[2];
    const float* q = xyz2 + (size_t)b * Mm * 3;
    float s = 0.f;
    for (int j = 0; j < Mm / 8; ++j) {
      const int m = i + j * 8;
      const float dx = x1 - q[m * 3 + 0];
      const float dy = y1 - q[m * 3 + 1];
      const float dz = z1 - q[m * 3 + 2];
      const float d  = sqrtf(dx * dx + dy * dy + dz * dz) + 1e-8f;
      const float inv = 1.0f / d;
      s += inv;
      wls[r * Mm + m] = f2bf(inv);
    }
    psum[t] = s;
  }
  __syncthreads();
  if (t < RT) {
    float s = 0.f;
    for (int q = 0; q < 8; ++q) s += psum[t * 8 + q];
    rowinv[t] = 1.0f / s;                    // normalization folded into epilogue
  }
  // ---- Phase 1b: points1 tile -> new_points columns [0,128) ----
  for (int idx = t; idx < RT * C1; idx += 512) {
    const int r = idx & 63, c = idx >> 6;
    npl[r * CIN + c] = f2bf(points1[((size_t)b * C1 + c) * Nn + n0 + r]);
  }
  __syncthreads();

  const int wave = t >> 5;                   // 16 waves
  const int lane = t & 31;
  const int col  = lane & 15;                // output col / A row (both lane&15)
  const int kh   = lane >> 4;                // K-half select

  // ---- Phase 2: interpolated = W[64x1024] @ points2^T[1024x256] ------------
  // wave -> one 16-col tile; B fragment reused across 4 row-tiles.
  {
    const unsigned short* bp = p2bf + ((size_t)b * C2 + wave * 16 + col) * Mm;
    v8f acc[4] = {v8f{}, v8f{}, v8f{}, v8f{}};
#pragma unroll 2
    for (int k0 = 0; k0 < Mm; k0 += 32) {
      const v16bf bb = __builtin_shufflevector(*(const v8bf*)(bp + k0 + kh * 8), *(const v8bf*)(bp + k0 + 16 + kh * 8), 0, 1, 2, 3, 4, 5, 6, 7, 8, 9, 10, 11, 12, 13, 14, 15);
#pragma unroll
      for (int rt = 0; rt < 4; ++rt) {
        union { v16bf v; v8bf h[2]; } a;
        const int ar = (rt * 16 + col) * Mm;
        a.h[0] = *(const v8bf*)&wls[ar + k0 + kh * 8];
        a.h[1] = *(const v8bf*)&wls[ar + k0 + 16 + kh * 8];
        acc[rt] = __builtin_amdgcn_wmma_f32_16x16x32_bf16(false, a.v, false, bb,
                                                          (short)0, acc[rt],
                                                          false, false);
      }
    }
#pragma unroll
    for (int rt = 0; rt < 4; ++rt)
#pragma unroll
      for (int e = 0; e < 8; ++e) {
        const int row = rt * 16 + e + 8 * kh;
        npl[row * CIN + C1 + wave * 16 + col] = f2bf(acc[rt][e] * rowinv[row]);
      }
  }
  __syncthreads();

  // ---- Phase 3: MLP1  h = relu(new_points[64x384] @ W1^T[384x256] + b1) ----
  {
    const unsigned short* wp = w1bf + (size_t)(wave * 16 + col) * CIN;
    v8f acc[4] = {v8f{}, v8f{}, v8f{}, v8f{}};
#pragma unroll 2
    for (int k0 = 0; k0 < CIN; k0 += 32) {
      const v16bf bb = __builtin_shufflevector(*(const v8bf*)(wp + k0 + kh * 8), *(const v8bf*)(wp + k0 + 16 + kh * 8), 0, 1, 2, 3, 4, 5, 6, 7, 8, 9, 10, 11, 12, 13, 14, 15);
#pragma unroll
      for (int rt = 0; rt < 4; ++rt) {
        union { v16bf v; v8bf h[2]; } a;
        const int ar = (rt * 16 + col) * CIN;
        a.h[0] = *(const v8bf*)&npl[ar + k0 + kh * 8];
        a.h[1] = *(const v8bf*)&npl[ar + k0 + 16 + kh * 8];
        acc[rt] = __builtin_amdgcn_wmma_f32_16x16x32_bf16(false, a.v, false, bb,
                                                          (short)0, acc[rt],
                                                          false, false);
      }
    }
    const float bias = b1[wave * 16 + col];
#pragma unroll
    for (int rt = 0; rt < 4; ++rt)
#pragma unroll
      for (int e = 0; e < 8; ++e) {
        const int row = rt * 16 + e + 8 * kh;
        hls[row * H1 + wave * 16 + col] = f2bf(fmaxf(acc[rt][e] + bias, 0.f));
      }
  }
  __syncthreads();

  // ---- Phase 4: MLP2  out = relu(h[64x256] @ W2^T[256x128] + b2) -----------
  // 8 col-tiles x 4 row-tiles = 32 tiles over 16 waves -> 2 row-tiles/wave.
  {
    const int o2  = (wave & 7) * 16 + col;
    const int rtb = (wave >> 3) * 2;
    const unsigned short* wp = w2bf + (size_t)o2 * H1;
    v8f acc[2] = {v8f{}, v8f{}};
#pragma unroll 2
    for (int k0 = 0; k0 < H1; k0 += 32) {
      const v16bf bb = __builtin_shufflevector(*(const v8bf*)(wp + k0 + kh * 8), *(const v8bf*)(wp + k0 + 16 + kh * 8), 0, 1, 2, 3, 4, 5, 6, 7, 8, 9, 10, 11, 12, 13, 14, 15);
#pragma unroll
      for (int q = 0; q < 2; ++q) {
        union { v16bf v; v8bf h[2]; } a;
        const int ar = ((rtb + q) * 16 + col) * H1;
        a.h[0] = *(const v8bf*)&hls[ar + k0 + kh * 8];
        a.h[1] = *(const v8bf*)&hls[ar + k0 + 16 + kh * 8];
        acc[q] = __builtin_amdgcn_wmma_f32_16x16x32_bf16(false, a.v, false, bb,
                                                         (short)0, acc[q],
                                                         false, false);
      }
    }
    const float bias = b2[o2];
#pragma unroll
    for (int q = 0; q < 2; ++q)
#pragma unroll
      for (int e = 0; e < 8; ++e) {
        const int row = (rtb + q) * 16 + e + 8 * kh;
        out[((size_t)b * H2 + o2) * Nn + n0 + row] = fmaxf(acc[q][e] + bias, 0.f);
      }
  }
}

extern "C" void kernel_launch(void* const* d_in, const int* in_sizes, int n_in,
                              void* d_out, int out_size, void* d_ws, size_t ws_size,
                              hipStream_t stream) {
  const float* xyz1    = (const float*)d_in[0];
  const float* xyz2    = (const float*)d_in[1];
  const float* points1 = (const float*)d_in[2];
  const float* points2 = (const float*)d_in[3];
  const float* W1      = (const float*)d_in[4];
  const float* b1      = (const float*)d_in[5];
  const float* W2      = (const float*)d_in[6];
  const float* b2      = (const float*)d_in[7];

  // workspace: bf16 copies of points2 / W1 / W2 (~8.65 MB total)
  unsigned short* p2bf = (unsigned short*)d_ws;
  unsigned short* w1bf = p2bf + (size_t)Bsz * C2 * Mm;   // +8,388,608 B
  unsigned short* w2bf = w1bf + (size_t)H1 * CIN;        // +196,608 B

  const int n_p2 = Bsz * C2 * Mm;
  const int n_w1 = H1 * CIN;
  const int n_w2 = H2 * H1;
  cvt_f32_bf16<<<(n_p2 + 255) / 256, 256, 0, stream>>>(points2, p2bf, n_p2);
  cvt_f32_bf16<<<(n_w1 + 255) / 256, 256, 0, stream>>>(W1, w1bf, n_w1);
  cvt_f32_bf16<<<(n_w2 + 255) / 256, 256, 0, stream>>>(W2, w2bf, n_w2);

  // one block per (batch, 64-row n-tile): 16 * 64 = 1024 blocks, 16 waves each
  fused_fp_kernel<<<dim3(Bsz * (Nn / RT)), dim3(512), LDS_TOTAL, stream>>>(
      xyz1, xyz2, points1, p2bf, w1bf, b1, w2bf, b2, (float*)d_out);
}
